// CNN_LSTM_24807731102148
// MI455X (gfx1250) — matrix operand order repaired and checked by reading, not compiled, not run on hardware
//
#include <hip/hip_runtime.h>
#include <hip/hip_bf16.h>
#include <math.h>

#define V_  10000
#define E_  512
#define H_  1024
#define L_  2
#define B_  64
#define T_  128
#define G4_ (4 * H_)   // 4096

typedef __attribute__((ext_vector_type(16))) __bf16 v16bf;
typedef __attribute__((ext_vector_type(8)))  float  v8f;

union FragBF { v16bf v; uint4 u[2]; };

__device__ __forceinline__ float sigf(float x) { return 1.0f / (1.0f + __expf(-x)); }

// A fragment (16x32 bf16): lane lr holds row m0+lr, chunks {lg*8..+7} and {+16..+23}
__device__ __forceinline__ void loadA(FragBF& f, const __bf16* p, int k0) {
    f.u[0] = *(const uint4*)(p + k0);
    f.u[1] = *(const uint4*)(p + k0 + 16);
}

// One per-lane 16B async copy: global -> LDS, tracked by ASYNCcnt, no VGPR data.
__device__ __forceinline__ void async_copy16(const __bf16* gsrc, const __bf16* ldst) {
    unsigned lds             = (unsigned)(size_t)ldst;
    unsigned long long gaddr = (unsigned long long)(size_t)gsrc;
    asm volatile("global_load_async_to_lds_b128 %0, %1, off"
                 :: "v"(lds), "v"(gaddr) : "memory");
}

// ---------------------------------------------------------------------------
// D(64 x 32-col group) = A1(64 x K) @ B1(N x K)^T [ + A2 @ B2^T if DUAL ]
// Block = 128 threads (4 waves = 4 m-tiles) sharing one 32-column n-group.
// Weight chunks (32 rows x 32 k, bf16) are staged into double-buffered LDS via
// global_load_async_to_lds_b128 and shared by all 4 waves (4x less L2 traffic);
// A fragments are register-pipelined direct loads. f32 accumulate via
// v_wmma_f32_16x16x32_bf16. K must be a power of two.
// ---------------------------------------------------------------------------
template <bool DUAL>
__global__ __launch_bounds__(128) void wmma_gemm_lds(
    const __bf16* __restrict__ A1, int lda1,
    const __bf16* __restrict__ B1,
    const __bf16* __restrict__ A2, int lda2,
    const __bf16* __restrict__ B2,
    int K,
    float*  __restrict__ Cf, int ldcf,
    __bf16* __restrict__ Cb, int ldcb)
{
    constexpr int NM = DUAL ? 2 : 1;
    __shared__ __bf16 smem[2][NM][32][32];   // [buffer][matrix][n_local][k]

    const int n0   = blockIdx.x << 5;        // 32-column group
    const int m0   = (threadIdx.x >> 5) << 4;
    const int lane = threadIdx.x & 31;
    const int lg   = lane >> 4;
    const int lr   = lane & 15;

    // async-copy mapping: thread tid moves 16B: row = tid>>2, quarter = tid&3
    const int crow  = threadIdx.x >> 2;
    const int cpart = (threadIdx.x & 3) * 8; // in bf16 elements
    const __bf16* g1 = B1 + (size_t)(n0 + crow) * K + cpart;
    const __bf16* g2 = DUAL ? B2 + (size_t)(n0 + crow) * K + cpart : (const __bf16*)0;

    const __bf16* a1row = A1 + (size_t)(m0 + lr) * lda1 + lg * 8;
    const __bf16* a2row = DUAL ? A2 + (size_t)(m0 + lr) * lda2 + lg * 8 : (const __bf16*)0;

    v8f acc[2];
#pragma unroll
    for (int j = 0; j < 2; ++j) acc[j] = (v8f){0.f,0.f,0.f,0.f,0.f,0.f,0.f,0.f};

    // prologue: stage chunk k=0 into buffer 0; A pipeline prologue
    async_copy16(g1, &smem[0][0][crow][cpart]);
    if (DUAL) async_copy16(g2, &smem[0][NM - 1][crow][cpart]);
    FragBF a1c, a2c;
    loadA(a1c, a1row, 0);
    if (DUAL) loadA(a2c, a2row, 0);

    const int kmask = K - 1;                 // K is a power of two (512 / 1024)
    int buf = 0;
    for (int k0 = 0; k0 < K; k0 += 32) {
        const int kn = (k0 + 32) & kmask;    // wrap-around keeps pipeline branchless

        // stage next chunk into the other buffer (ASYNCcnt += NM per thread)
        async_copy16(g1 + kn, &smem[buf ^ 1][0][crow][cpart]);
        if (DUAL) async_copy16(g2 + kn, &smem[buf ^ 1][NM - 1][crow][cpart]);

        // next A fragments (register double-buffer)
        FragBF a1n, a2n;
        loadA(a1n, a1row, kn);
        if (DUAL) loadA(a2n, a2row, kn);

        // wait for my previous copies (chunk in smem[buf]); barrier -> all done
        if (DUAL) asm volatile("s_wait_asynccnt 0x2" ::: "memory");
        else      asm volatile("s_wait_asynccnt 0x1" ::: "memory");
        __syncthreads();

        // B fragments from LDS: lane = column j*16+lr, k chunks {lg*8..+7} and {+16..+23}
        FragBF b1f[2], b2f[2];
#pragma unroll
        for (int j = 0; j < 2; ++j) {
            b1f[j].u[0] = *(const uint4*)&smem[buf][0][j * 16 + lr][lg * 8];
            b1f[j].u[1] = *(const uint4*)&smem[buf][0][j * 16 + lr][lg * 8 + 16];
            if (DUAL) {
                b2f[j].u[0] = *(const uint4*)&smem[buf][NM - 1][j * 16 + lr][lg * 8];
                b2f[j].u[1] = *(const uint4*)&smem[buf][NM - 1][j * 16 + lr][lg * 8 + 16];
            }
        }

#pragma unroll
        for (int j = 0; j < 2; ++j)
            acc[j] = __builtin_amdgcn_wmma_f32_16x16x32_bf16(
                         false, a1c.v, false, b1f[j].v, (short)0, acc[j], false, false);
        if (DUAL) {
#pragma unroll
            for (int j = 0; j < 2; ++j)
                acc[j] = __builtin_amdgcn_wmma_f32_16x16x32_bf16(
                             false, a2c.v, false, b2f[j].v, (short)0, acc[j], false, false);
        }

        __syncthreads();                     // everyone done reading smem[buf]
        a1c = a1n;
        if (DUAL) a2c = a2n;
        buf ^= 1;
    }

    // --- store epilogue: one branch per destination, strength-reduced rows ---
    // C/D layout: VGPR i -> M = m0 + i + 8*lg, N = n0 + j*16 + lr
    const int rbase = m0 + (lg << 3);
    if (Cf) {
        float* p = Cf + (size_t)rbase * ldcf + n0 + lr;
#pragma unroll
        for (int j = 0; j < 2; ++j) {
            float* pj = p + j * 16;
#pragma unroll
            for (int i = 0; i < 8; ++i)
                pj[(size_t)i * ldcf] = acc[j][i];
        }
    }
    if (Cb) {
        __bf16* p = Cb + (size_t)rbase * ldcb + n0 + lr;
#pragma unroll
        for (int j = 0; j < 2; ++j) {
            __bf16* pj = p + j * 16;
#pragma unroll
            for (int i = 0; i < 8; ++i)
                pj[(size_t)i * ldcb] = (__bf16)acc[j][i];
        }
    }
}

// ---------------------------------------------------------------------------
// LSTM cell pointwise: gates (B x 4H) f32 + fused bias -> c update,
// act = sigmoid(o) * tanh(c_new)
// ---------------------------------------------------------------------------
__global__ __launch_bounds__(256) void lstm_pointwise(
    const float* __restrict__ gates,
    const float* __restrict__ bsum,
    float*  __restrict__ c,
    __bf16* __restrict__ act)
{
    const int idx = blockIdx.x * blockDim.x + threadIdx.x;   // B_*H_
    if (idx >= B_ * H_) return;
    const int b = idx / H_;
    const int h = idx - b * H_;
    const float* g = gates + (size_t)b * G4_;

    const float gi = sigf (g[h]           + bsum[h]);
    const float gf = sigf (g[H_ + h]      + bsum[H_ + h]);
    const float gg = tanhf(g[2 * H_ + h]  + bsum[2 * H_ + h]);
    const float go = sigf (g[3 * H_ + h]  + bsum[3 * H_ + h]);

    const float cn = gf * c[idx] + gi * gg;
    c[idx]   = cn;
    act[idx] = (__bf16)(go * tanhf(cn));
}

// ---------------------------------------------------------------------------
// seq[b, 0, :] = images[b, 0, :]; seq[b, t, :] = embed[captions[b, t-1], :]
// ---------------------------------------------------------------------------
__global__ __launch_bounds__(256) void build_seq(
    const float* __restrict__ images,
    const int*   __restrict__ captions,
    const float* __restrict__ embed,
    __bf16* __restrict__ seq)
{
    const int idx = blockIdx.x * blockDim.x + threadIdx.x;  // B_*T_*E_
    if (idx >= B_ * T_ * E_) return;
    const int e = idx & (E_ - 1);
    const int r = idx / E_;         // b*T + t
    const int t = r & (T_ - 1);
    const int b = r / T_;
    float v;
    if (t == 0) v = images[(size_t)b * E_ + e];
    else        v = embed[(size_t)captions[b * T_ + (t - 1)] * E_ + e];
    seq[idx] = (__bf16)v;
}

__global__ __launch_bounds__(256) void f32_to_bf16(
    const float* __restrict__ in, __bf16* __restrict__ out, int n)
{
    const int i = blockIdx.x * blockDim.x + threadIdx.x;
    if (i < n) out[i] = (__bf16)in[i];
}

__global__ __launch_bounds__(256) void sum_bias(
    const float* __restrict__ a, const float* __restrict__ b,
    float* __restrict__ out, int n)
{
    const int i = blockIdx.x * blockDim.x + threadIdx.x;
    if (i < n) out[i] = a[i] + b[i];
}

__global__ __launch_bounds__(256) void zero_state(
    __bf16* __restrict__ h, int nh, float* __restrict__ c, int nc)
{
    const int i = blockIdx.x * blockDim.x + threadIdx.x;
    if (i < nh) h[i] = (__bf16)0.f;
    if (i < nc) c[i] = 0.f;
}

// ---------------------------------------------------------------------------
// In-place softmax over last dim (E_=512), one 256-thread block per row.
// ---------------------------------------------------------------------------
__global__ __launch_bounds__(256) void softmax_rows(float* __restrict__ x)
{
    __shared__ float red[256];
    float* p = x + (size_t)blockIdx.x * E_;
    const int tid = threadIdx.x;

    float v0 = p[tid], v1 = p[tid + 256];
    float m = fmaxf(v0, v1);
    red[tid] = m;
    __syncthreads();
    for (int s = 128; s > 0; s >>= 1) {
        if (tid < s) red[tid] = fmaxf(red[tid], red[tid + s]);
        __syncthreads();
    }
    m = red[0];
    __syncthreads();

    float e0 = __expf(v0 - m), e1 = __expf(v1 - m);
    red[tid] = e0 + e1;
    __syncthreads();
    for (int s = 128; s > 0; s >>= 1) {
        if (tid < s) red[tid] += red[tid + s];
        __syncthreads();
    }
    const float inv = 1.0f / red[0];
    p[tid]       = e0 * inv;
    p[tid + 256] = e1 * inv;
}

// ---------------------------------------------------------------------------
extern "C" void kernel_launch(void* const* d_in, const int* in_sizes, int n_in,
                              void* d_out, int out_size, void* d_ws, size_t ws_size,
                              hipStream_t stream)
{
    (void)in_sizes; (void)n_in; (void)out_size; (void)ws_size;

    const float* images   = (const float*)d_in[0];
    const int*   captions = (const int*)  d_in[1];
    const float* embed    = (const float*)d_in[2];
    const float* W_ih     = (const float*)d_in[3];
    const float* W_hh     = (const float*)d_in[4];
    const float* W_hr     = (const float*)d_in[5];
    const float* b_ih     = (const float*)d_in[6];
    const float* b_hh     = (const float*)d_in[7];
    float* out = (float*)d_out;

    // ---- workspace partition (256B aligned) ----
    uintptr_t base = (uintptr_t)d_ws;
    auto alloc = [&](size_t bytes) {
        base = (base + 255) & ~(uintptr_t)255;
        uintptr_t p = base; base += bytes; return (void*)p;
    };
    const size_t nWih = (size_t)L_ * G4_ * E_;   // 4,194,304
    const size_t nWhh = nWih;
    const size_t nWhr = (size_t)L_ * E_ * H_;    // 1,048,576
    const size_t nSeq = (size_t)B_ * T_ * E_;    // 4,194,304
    __bf16* Wih_bf = (__bf16*)alloc(nWih * 2);
    __bf16* Whh_bf = (__bf16*)alloc(nWhh * 2);
    __bf16* Whr_bf = (__bf16*)alloc(nWhr * 2);
    __bf16* seq_bf = (__bf16*)alloc(nSeq * 2);
    __bf16* h_bf   = (__bf16*)alloc((size_t)L_ * B_ * E_ * 2);
    float*  c_f    = (float*) alloc((size_t)L_ * B_ * H_ * 4);
    float*  gates  = (float*) alloc((size_t)B_ * G4_ * 4);
    __bf16* act    = (__bf16*)alloc((size_t)B_ * H_ * 2);
    float*  bsum   = (float*) alloc((size_t)L_ * G4_ * 4);

    // ---- one-time prep ----
    f32_to_bf16<<<(int)(nWih / 256), 256, 0, stream>>>(W_ih, Wih_bf, (int)nWih);
    f32_to_bf16<<<(int)(nWhh / 256), 256, 0, stream>>>(W_hh, Whh_bf, (int)nWhh);
    f32_to_bf16<<<(int)(nWhr / 256), 256, 0, stream>>>(W_hr, Whr_bf, (int)nWhr);
    build_seq<<<(int)(nSeq / 256), 256, 0, stream>>>(images, captions, embed, seq_bf);
    sum_bias<<<(L_ * G4_) / 256, 256, 0, stream>>>(b_ih, b_hh, bsum, L_ * G4_);
    zero_state<<<512, 256, 0, stream>>>(h_bf, L_ * B_ * E_, c_f, L_ * B_ * H_);

    // ---- sequential LSTM scan ----
    const int gate_blks = G4_ / 32;      // 128 blocks x 128 thr (4 m-waves each)
    const int proj_blks = E_ / 32;       // 16 blocks
    const int pw_blks   = (B_ * H_) / 256;

    for (int t = 0; t < T_; ++t) {
        for (int l = 0; l < L_; ++l) {
            const __bf16* inp; int ldin;
            if (l == 0) { inp = seq_bf + (size_t)t * E_; ldin = T_ * E_; }
            else        { inp = h_bf;                    ldin = E_;      }
            __bf16* hl = h_bf + (size_t)l * B_ * E_;

            // gates = inp @ W_ih[l]^T + h[l] @ W_hh[l]^T
            wmma_gemm_lds<true><<<gate_blks, 128, 0, stream>>>(
                inp, ldin, Wih_bf + (size_t)l * G4_ * E_,
                hl, E_,    Whh_bf + (size_t)l * G4_ * E_,
                E_,
                gates, G4_, (__bf16*)0, 0);

            // cell update + activation
            lstm_pointwise<<<pw_blks, 256, 0, stream>>>(
                gates, bsum + (size_t)l * G4_,
                c_f + (size_t)l * B_ * H_, act);

            // h_new = act @ W_hr[l]^T  (bf16 state always; f32 logits on layer 1)
            float* logit = (l == L_ - 1) ? (out + (size_t)t * E_) : (float*)0;
            wmma_gemm_lds<false><<<proj_blks, 128, 0, stream>>>(
                act, H_, Whr_bf + (size_t)l * E_ * H_,
                (const __bf16*)0, 0, (const __bf16*)0,
                H_,
                logit, T_ * E_, hl, E_);
        }
    }

    // ---- softmax over E per (b, t) row ----
    softmax_rows<<<B_ * T_, 256, 0, stream>>>(out);
}
